// LSTM_75299366634042
// MI455X (gfx1250) — matrix operand order repaired and checked by reading, not compiled, not run on hardware
//
#include <hip/hip_runtime.h>

typedef __attribute__((ext_vector_type(16))) _Float16 v16h;
typedef __attribute__((ext_vector_type(8)))  _Float16 v8h;
typedef __attribute__((ext_vector_type(8)))  float    v8f;

#define BB 64
#define SS 1024
#define II 256
#define HH 256
#define NG 1024  // 4*HH, gates stacked i,f,g,o

// ---------------------------------------------------------------------------
// Kernel 0: pack weights to f16 [4H x K] row-major, combine input+hidden bias
// ---------------------------------------------------------------------------
__global__ void lstm_pack_kernel(
    const float* __restrict__ w_ii, const float* __restrict__ w_if,
    const float* __restrict__ w_ig, const float* __restrict__ w_io,
    const float* __restrict__ w_hi, const float* __restrict__ w_hf,
    const float* __restrict__ w_hg, const float* __restrict__ w_ho,
    const float* __restrict__ b_ii, const float* __restrict__ b_if,
    const float* __restrict__ b_ig, const float* __restrict__ b_io,
    const float* __restrict__ b_hi, const float* __restrict__ b_hf,
    const float* __restrict__ b_hg, const float* __restrict__ b_ho,
    _Float16* __restrict__ Wx, _Float16* __restrict__ Wh,
    float* __restrict__ bsum)
{
    const int WN = NG * II;  // 262144
    int idx = blockIdx.x * blockDim.x + threadIdx.x;
    if (idx < WN) {
        int row = idx >> 8, k = idx & 255;
        int g = row >> 8, n = row & 255;
        const float* w = (g == 0) ? w_ii : (g == 1) ? w_if : (g == 2) ? w_ig : w_io;
        Wx[idx] = (_Float16)w[n * II + k];
    } else if (idx < 2 * WN) {
        int j = idx - WN;
        int row = j >> 8, k = j & 255;
        int g = row >> 8, n = row & 255;
        const float* w = (g == 0) ? w_hi : (g == 1) ? w_hf : (g == 2) ? w_hg : w_ho;
        Wh[j] = (_Float16)w[n * HH + k];
    } else if (idx < 2 * WN + NG) {
        int j = idx - 2 * WN;
        int g = j >> 8, n = j & 255;
        const float* bi = (g == 0) ? b_ii : (g == 1) ? b_if : (g == 2) ? b_ig : b_io;
        const float* bh = (g == 0) ? b_hi : (g == 1) ? b_hf : (g == 2) ? b_hg : b_ho;
        bsum[j] = bi[n] + bh[n];
    }
}

// ---------------------------------------------------------------------------
// Kernel 1: input projections  X[s][b][4H] = inputs[:, s, :] @ Wx^T + bsum
// One block per s. A-panel (64x256 f32) staged ONCE via async global->LDS,
// converted to f16 in LDS, then all 8 waves feed WMMA A-fragments from LDS.
// mt/q kept as real loops so the scheduler cannot hoist+spill B-fragments.
// ---------------------------------------------------------------------------
__global__ void lstm_xproj_kernel(const float* __restrict__ inp,     // [B,S,I]
                                  const _Float16* __restrict__ Wx,   // [4H,I]
                                  const float* __restrict__ bsum,    // [4H]
                                  _Float16* __restrict__ X)          // [S,B,4H]
{
    extern __shared__ char xsm[];
    float*    Af = (float*)xsm;                       // [64][256] f32, 64KB
    _Float16* Ah = (_Float16*)(xsm + 65536);          // [64][256] f16, 32KB

    const int s = blockIdx.x;
    const unsigned tid  = threadIdx.x;
    const unsigned lane = tid & 31u;
    const unsigned wave = tid >> 5;   // 0..7
    const unsigned lh = lane & 15u;
    const unsigned kg = lane >> 4;

    // ---- stage A-panel: async global -> LDS, 64 floats (16 x b128) / thread
    {
        const unsigned base = tid * 64u;              // flat f32 index
        const unsigned brow = base >> 8;              // batch row 0..63
        const unsigned k0   = base & 255u;
        const float* gp = inp + ((size_t)brow * SS + s) * II + k0;
        const unsigned lds0 = (unsigned)(size_t)(void*)(Af + base);
        #pragma unroll
        for (int i = 0; i < 16; ++i) {
            unsigned lo = lds0 + (unsigned)(i * 16);
            unsigned long long ga = (unsigned long long)(size_t)(gp + i * 4);
            asm volatile("global_load_async_to_lds_b128 %0, %1, off"
                         :: "v"(lo), "v"(ga) : "memory");
        }
        asm volatile("s_wait_asynccnt 0" ::: "memory");
    }
    __syncthreads();
    // ---- convert staged panel to f16 (one conversion per element total)
    {
        const unsigned base = tid * 64u;
        #pragma unroll
        for (int i = 0; i < 64; ++i) Ah[base + i] = (_Float16)Af[base + i];
    }
    __syncthreads();

    // ---- WMMA: wave owns N-tiles 8w..8w+7
    _Float16* xp = X + (size_t)s * (BB * NG);
    #pragma unroll 1
    for (int mt = 0; mt < 4; ++mt) {
        #pragma unroll 1
        for (int q = 0; q < 8; ++q) {
            const int ncol = (wave * 8 + q) * 16 + lh;
            v8f acc = {};
            #pragma unroll
            for (int kc = 0; kc < 8; ++kc) {
                // A fragment: M = mt*16+lh, K = kc*32 + kg*16 + {0..15}
                const _Float16* hp = Ah + (mt * 16 + lh) * II + kc * 32 + kg * 16;
                v8h alo = *(const v8h*)(hp);
                v8h ahi = *(const v8h*)(hp + 8);
                v16h a;
                #pragma unroll
                for (int j = 0; j < 8; ++j) { a[j] = alo[j]; a[j + 8] = ahi[j]; }
                v16h bf = *(const v16h*)(Wx + (size_t)ncol * II + kc * 32 + kg * 16);
                acc = __builtin_amdgcn_wmma_f32_16x16x32_f16(
                    false, a, false, bf, (short)0, acc, false, false);
            }
            const float bias = bsum[ncol];
            #pragma unroll
            for (int r = 0; r < 8; ++r) {
                int brow = mt * 16 + kg * 8 + r;
                xp[(size_t)brow * NG + ncol] = (_Float16)(acc[r] + bias);
            }
        }
    }
}

// ---------------------------------------------------------------------------
// Kernel 2: persistent single-workgroup scan. 1024 threads = 32 waves.
// h(f16) + gates(f32) resident in 288KB LDS; c in registers; W_h streamed
// from L2 (512KB, stays hot) each step; X[t] loads hoisted above the GEMM.
// kc kept as a (x2-unrolled) real loop to bound live registers (no spills).
// ---------------------------------------------------------------------------
__global__ void lstm_scan_kernel(const _Float16* __restrict__ X,   // [S,B,4H]
                                 const _Float16* __restrict__ Wh,  // [4H,H]
                                 const float* __restrict__ h0,     // [B,H]
                                 const float* __restrict__ c0,     // [B,H]
                                 float* __restrict__ out)          // [B,S,H] ++ hN ++ cN
{
    extern __shared__ char smem[];
    _Float16* hl = (_Float16*)smem;                                     // [B][H]  32KB
    float*    gl = (float*)(smem + (size_t)BB * HH * sizeof(_Float16)); // [B][4H] 256KB

    const unsigned tid  = threadIdx.x;
    const unsigned lane = tid & 31u;
    const unsigned wave = tid >> 5;     // 0..31, owns N-tiles {2w, 2w+1}
    const unsigned lh = lane & 15u;
    const unsigned kg = lane >> 4;

    // init h (LDS f16) and c (regs): thread owns 16 consecutive (b,h)
    float c[16];
    const int fi = tid * 16;
    const int bo = fi >> 8;
    const int hb = fi & 255;
    #pragma unroll
    for (int j = 0; j < 16; ++j) {
        c[j] = c0[fi + j];
        hl[bo * HH + hb + j] = (_Float16)h0[fi + j];
    }
    __syncthreads();

    const int n0 = (wave * 2 + 0) * 16 + lh;   // row of Wh for N-tile 0
    const int n1 = (wave * 2 + 1) * 16 + lh;   // row of Wh for N-tile 1

    for (int t = 0; t < SS; ++t) {
        // ---- issue X[t] loads early: latency overlaps the WMMA chain ----
        const _Float16* xt = X + (size_t)t * (BB * NG) + bo * NG;
        v16h xi = *(const v16h*)(xt + 0 * HH + hb);
        v16h xf = *(const v16h*)(xt + 1 * HH + hb);
        v16h xg = *(const v16h*)(xt + 2 * HH + hb);
        v16h xo = *(const v16h*)(xt + 3 * HH + hb);
        if (t + 1 < SS)   // warm L2/L0 for the next step
            __builtin_prefetch(X + (size_t)(t + 1) * (BB * NG) + bo * NG + hb, 0, 1);

        // ---- gates = h @ Wh^T : 64 WMMAs per wave ----
        v8f acc[2][4];
        #pragma unroll
        for (int q = 0; q < 2; ++q)
            #pragma unroll
            for (int mt = 0; mt < 4; ++mt) acc[q][mt] = (v8f){};

        #pragma unroll 2
        for (int kc = 0; kc < 8; ++kc) {
            // B fragments for this K-chunk (hot in L2)
            v16h bf0 = *(const v16h*)(Wh + (size_t)n0 * HH + kc * 32 + kg * 16);
            v16h bf1 = *(const v16h*)(Wh + (size_t)n1 * HH + kc * 32 + kg * 16);
            #pragma unroll
            for (int mt = 0; mt < 4; ++mt) {
                const _Float16* hp = hl + (mt * 16 + lh) * HH + kc * 32 + kg * 16;
                v8h alo = *(const v8h*)(hp);
                v8h ahi = *(const v8h*)(hp + 8);
                v16h a;
                #pragma unroll
                for (int j = 0; j < 8; ++j) { a[j] = alo[j]; a[j + 8] = ahi[j]; }
                acc[0][mt] = __builtin_amdgcn_wmma_f32_16x16x32_f16(
                    false, a, false, bf0, (short)0, acc[0][mt], false, false);
                acc[1][mt] = __builtin_amdgcn_wmma_f32_16x16x32_f16(
                    false, a, false, bf1, (short)0, acc[1][mt], false, false);
            }
        }
        #pragma unroll
        for (int q = 0; q < 2; ++q) {
            const int ncol = (wave * 2 + q) * 16 + lh;
            #pragma unroll
            for (int mt = 0; mt < 4; ++mt)
                #pragma unroll
                for (int r = 0; r < 8; ++r)
                    gl[(mt * 16 + kg * 8 + r) * NG + ncol] = acc[q][mt][r];
        }
        __syncthreads();

        // ---- elementwise LSTM cell update ----
        {
            const float* gi = gl + bo * NG + 0 * HH + hb;
            const float* gf = gl + bo * NG + 1 * HH + hb;
            const float* gg = gl + bo * NG + 2 * HH + hb;
            const float* go = gl + bo * NG + 3 * HH + hb;
            float* orow = out + ((size_t)bo * SS + t) * HH + hb;

            float hv[16];
            #pragma unroll
            for (int j = 0; j < 16; ++j) {
                float iv = gi[j] + (float)xi[j];
                float fv = gf[j] + (float)xf[j];
                float gv = gg[j] + (float)xg[j];
                float ov = go[j] + (float)xo[j];
                float i_t = 1.f / (1.f + __expf(-iv));
                float f_t = 1.f / (1.f + __expf(-fv));
                float g_t = tanhf(gv);
                float o_t = 1.f / (1.f + __expf(-ov));
                c[j] = f_t * c[j] + i_t * g_t;
                hv[j] = o_t * tanhf(c[j]);
                orow[j] = hv[j];
                hl[bo * HH + hb + j] = (_Float16)hv[j];
            }
            if (t == SS - 1) {
                float* hN = out + (size_t)BB * SS * HH;
                float* cN = hN + BB * HH;
                #pragma unroll
                for (int j = 0; j < 16; ++j) { hN[fi + j] = hv[j]; cN[fi + j] = c[j]; }
            }
        }
        __syncthreads();
    }
}

// ---------------------------------------------------------------------------
extern "C" void kernel_launch(void* const* d_in, const int* in_sizes, int n_in,
                              void* d_out, int out_size, void* d_ws, size_t ws_size,
                              hipStream_t stream) {
    const float* inputs = (const float*)d_in[0];
    const float* h0     = (const float*)d_in[1];
    const float* c0     = (const float*)d_in[2];
    const float* w_ii = (const float*)d_in[3];
    const float* w_if = (const float*)d_in[4];
    const float* w_ig = (const float*)d_in[5];
    const float* w_io = (const float*)d_in[6];
    const float* b_ii = (const float*)d_in[7];
    const float* b_if = (const float*)d_in[8];
    const float* b_ig = (const float*)d_in[9];
    const float* b_io = (const float*)d_in[10];
    const float* w_hi = (const float*)d_in[11];
    const float* w_hf = (const float*)d_in[12];
    const float* w_hg = (const float*)d_in[13];
    const float* w_ho = (const float*)d_in[14];
    const float* b_hi = (const float*)d_in[15];
    const float* b_hf = (const float*)d_in[16];
    const float* b_hg = (const float*)d_in[17];
    const float* b_ho = (const float*)d_in[18];
    float* out = (float*)d_out;

    char* ws = (char*)d_ws;
    _Float16* Wx   = (_Float16*)(ws);                 // 4H*I f16 = 512 KB
    _Float16* Wh   = (_Float16*)(ws + 524288);        // 4H*H f16 = 512 KB
    float*    bsum = (float*)(ws + 1048576);          // 4H f32   =   4 KB
    _Float16* X    = (_Float16*)(ws + 1052672);       // S*B*4H f16 = 128 MB

    // Phase 0: pack weights/biases
    {
        int total = 2 * NG * II + NG;
        int blocks = (total + 255) / 256;
        lstm_pack_kernel<<<blocks, 256, 0, stream>>>(
            w_ii, w_if, w_ig, w_io, w_hi, w_hf, w_hg, w_ho,
            b_ii, b_if, b_ig, b_io, b_hi, b_hf, b_hg, b_ho,
            Wx, Wh, bsum);
    }
    // Phase 1: big input-projection GEMM (async-LDS staged A, WMMA)
    lstm_xproj_kernel<<<SS, 256, 96 * 1024, stream>>>(inputs, Wx, bsum, X);

    // Phase 2: persistent single-WGP recurrent scan (288 KB LDS)
    size_t smem = (size_t)BB * HH * sizeof(_Float16) + (size_t)BB * NG * sizeof(float);
    lstm_scan_kernel<<<1, 1024, smem, stream>>>(X, Wh, h0, c0, out);
}
